// SelfAttention_4844723110323
// MI455X (gfx1250) — compile-verified, matrix operand order checked by reading, not run on hardware
//
#include <hip/hip_runtime.h>
#include <hip/hip_bf16.h>

#define EMBED 1024
#define HEADS 16
#define HDIM  64
#define BATCH 4
#define SEQ   2048

typedef __attribute__((ext_vector_type(16))) __bf16        v16bf;
typedef __attribute__((ext_vector_type(8)))  float         v8f;
typedef __attribute__((ext_vector_type(4)))  unsigned int  v4u;
typedef __attribute__((ext_vector_type(8)))  int           v8i;
typedef __attribute__((ext_vector_type(4)))  int           v4i;

#define DEVINL __device__ __forceinline__

// K index striping for 16-bit A (16x32) and B (32x16) fragments (wave32):
// lanes 0-15 hold K 0-7 (vgpr pairs 0-3) and 16-23 (pairs 4-7); lanes 16-31 hold 8-15 / 24-31.
DEVINL int frag_kbase(int grp, int j) {
    return (j < 4) ? ((grp ? 8 : 0) + 2 * j) : ((grp ? 24 : 16) + 2 * (j - 4));
}

// A fragment (M=lane&15 rows) or B fragment stored N-major (row n = lane&15, K contiguous).
DEVINL v16bf load_frag_rowmajor_bf16(const __bf16* src, int ld) {
    const int lane = threadIdx.x & 31;
    const int r = lane & 15, grp = lane >> 4;
    v16bf f;
#pragma unroll
    for (int j = 0; j < 8; ++j) {
        const int kb = frag_kbase(grp, j);
        f[2 * j]     = src[(size_t)r * ld + kb];
        f[2 * j + 1] = src[(size_t)r * ld + kb + 1];
    }
    return f;
}

DEVINL v16bf load_frag_rowmajor_f32(const float* src, int ld) {
    const int lane = threadIdx.x & 31;
    const int r = lane & 15, grp = lane >> 4;
    v16bf f;
#pragma unroll
    for (int j = 0; j < 8; ++j) {
        const int kb = frag_kbase(grp, j);
        f[2 * j]     = (__bf16)src[(size_t)r * ld + kb];
        f[2 * j + 1] = (__bf16)src[(size_t)r * ld + kb + 1];
    }
    return f;
}

// B fragment stored K-major: element (k, n) at src[k*ld + n].
DEVINL v16bf load_frag_kmajor_bf16(const __bf16* src, int ld) {
    const int lane = threadIdx.x & 31;
    const int n = lane & 15, grp = lane >> 4;
    v16bf f;
#pragma unroll
    for (int j = 0; j < 8; ++j) {
        const int kb = frag_kbase(grp, j);
        f[2 * j]     = src[(size_t)kb * ld + n];
        f[2 * j + 1] = src[(size_t)(kb + 1) * ld + n];
    }
    return f;
}

DEVINL v8f wmma_bf16(v16bf a, v16bf b, v8f c) {
    return __builtin_amdgcn_wmma_f32_16x16x32_bf16(false, a, false, b, (short)0, c,
                                                   false, false);
}

// ---------------------------------------------------------------------------
// TDM: async 2D tile load Global -> LDS (Tensor Data Mover, tracked TENSORcnt).
// Descriptor packing per CDNA5 ISA ch.8 (D# group0/group1; groups 2/3 zero for 2D).
// dims/tiles/stride in elements (data_size = 2 bytes, bf16); addresses in bytes.
// This toolchain exposes the 6-arg builtin form (extra int32x8 group).
// ---------------------------------------------------------------------------
DEVINL void tdm_load_2d_bf16(unsigned lds_addr, const void* gaddr,
                             unsigned dim0, unsigned dim1,
                             unsigned tile0, unsigned tile1, unsigned stride0) {
    const unsigned long long ga = (unsigned long long)(uintptr_t)gaddr;
    v4u g0;
    g0[0] = 1u;                                        // count=1, is_restore=0, no gather
    g0[1] = lds_addr;                                  // lds_addr [63:32]
    g0[2] = (unsigned)ga;                              // global_addr [95:64]
    g0[3] = (unsigned)(ga >> 32) | (2u << 30);         // global_addr[56:32] | type=2
    v8i g1;
    g1[0] = (int)0x00010000u;                          // wg_mask=0, data_size=1 (2B)
    g1[1] = (int)((dim0 & 0xFFFFu) << 16);             // tensor_dim0[15:0] @ bits 63:48
    g1[2] = (int)((dim0 >> 16) | ((dim1 & 0xFFFFu) << 16));  // dim0 hi | dim1 lo
    g1[3] = (int)((dim1 >> 16) | (tile0 << 16));       // dim1 hi | tile_dim0
    g1[4] = (int)tile1;                                // tile_dim1 (tile_dim2 = 0)
    g1[5] = (int)stride0;                              // tensor_dim0_stride[31:0]
    g1[6] = 0;                                         // stride hi | dim1_stride lo
    g1[7] = 0;
    v4i gz4;
    gz4[0] = 0; gz4[1] = 0; gz4[2] = 0; gz4[3] = 0;
    v8i gz8;
    gz8[0] = 0; gz8[1] = 0; gz8[2] = 0; gz8[3] = 0;
    gz8[4] = 0; gz8[5] = 0; gz8[6] = 0; gz8[7] = 0;
    __builtin_amdgcn_tensor_load_to_lds(g0, g1, gz4, gz4, gz8, 0);
}

// ---------------------------------------------------------------------------
// Kernel 1: per-head projection  out[n,h,s,e'] = sum_d x[n,s,h*64+d] * W[e',d]
// One wave per 16-row tile; output bf16 in [B,H,S,64].
// ---------------------------------------------------------------------------
__global__ void proj_kernel(const float* __restrict__ x, const float* __restrict__ W,
                            __bf16* __restrict__ out) {
    const int wave = threadIdx.x >> 5;
    const int tile = blockIdx.x * (blockDim.x >> 5) + wave;  // 0 .. B*H*S/16-1
    const int st = tile & (SEQ / 16 - 1);
    const int nh = tile >> 7;           // SEQ/16 == 128
    const int h = nh & (HEADS - 1);
    const int n = nh >> 4;
    const int s0 = st * 16;

    const int lane = threadIdx.x & 31;
    const int r = lane & 15, grp = lane >> 4;

    const float* xa = x + ((size_t)n * SEQ + s0) * EMBED + h * HDIM;
    v16bf a0 = load_frag_rowmajor_f32(xa, EMBED);        // K = d 0..31
    v16bf a1 = load_frag_rowmajor_f32(xa + 32, EMBED);   // K = d 32..63

    v8f acc[4] = {};
#pragma unroll
    for (int nc = 0; nc < 4; ++nc) {
        const float* wb = W + (size_t)(nc * 16) * HDIM;  // B(k=d, n=e') = W[e',d] (N-major)
        v16bf b0 = load_frag_rowmajor_f32(wb, HDIM);
        v16bf b1 = load_frag_rowmajor_f32(wb + 32, HDIM);
        acc[nc] = wmma_bf16(a0, b0, acc[nc]);
        acc[nc] = wmma_bf16(a1, b1, acc[nc]);
    }

    __bf16* o = out + (((size_t)n * HEADS + h) * SEQ + s0) * HDIM;
#pragma unroll
    for (int nc = 0; nc < 4; ++nc)
#pragma unroll
        for (int i = 0; i < 8; ++i) {
            const int M = grp * 8 + i;
            o[(size_t)M * HDIM + nc * 16 + r] = (__bf16)acc[nc][i];
        }
}

// ---------------------------------------------------------------------------
// Kernel 2: flash attention per (n, h). Block = 8 waves x 16 q-rows = 128 rows.
// K/V tiles staged into LDS once per block by the TDM (wave 0 issues
// tensor_load_to_lds, waits TENSORcnt, block barrier) -> 8x less global traffic.
// Online softmax; P staged through per-wave LDS to reshape C-layout ->
// A-fragment layout for the P @ V matmul.
// ---------------------------------------------------------------------------
__global__ void attn_kernel(const __bf16* __restrict__ Qp, const __bf16* __restrict__ Kp,
                            const __bf16* __restrict__ Vp, const int* __restrict__ mask,
                            __bf16* __restrict__ attn_out) {
    __shared__ __bf16 ktile[32 * HDIM];   // 4 KB: keys kt..kt+31, row-major [key][d]
    __shared__ __bf16 vtile[32 * HDIM];   // 4 KB
    __shared__ __bf16 p_lds[8][16 * 32];  // 8 KB: per-wave P reshape buffer

    const int wave = threadIdx.x >> 5;
    const int lane = threadIdx.x & 31;
    const int r = lane & 15, grp = lane >> 4;

    const int qb = blockIdx.x & (SEQ / 128 - 1);  // 16 q-blocks per (n,h)
    const int nh = blockIdx.x >> 4;
    const int h = nh & (HEADS - 1);
    const int n = nh >> 4;
    const int q0 = qb * 128 + wave * 16;

    const __bf16* qtile = Qp + (((size_t)n * HEADS + h) * SEQ + q0) * HDIM;
    const __bf16* kbase = Kp + ((size_t)n * HEADS + h) * SEQ * HDIM;
    const __bf16* vbase = Vp + ((size_t)n * HEADS + h) * SEQ * HDIM;

    const unsigned klds = (unsigned)(uintptr_t)&ktile[0];
    const unsigned vlds = (unsigned)(uintptr_t)&vtile[0];

    v16bf aq0 = load_frag_rowmajor_bf16(qtile, HDIM);
    v16bf aq1 = load_frag_rowmajor_bf16(qtile + 32, HDIM);

    v8f oacc[4] = {};
    float mrow[8], lrow[8];
#pragma unroll
    for (int i = 0; i < 8; ++i) { mrow[i] = -1.0e30f; lrow[i] = 0.0f; }

    const float scale = 0.03125f;  // 1/sqrt(EMBED)

    for (int kt = 0; kt < SEQ; kt += 32) {
        __syncthreads();  // previous iteration done reading ktile/vtile
        if (wave == 0) {
            tdm_load_2d_bf16(klds, kbase + (size_t)kt * HDIM, HDIM, 32, HDIM, 32, HDIM);
            tdm_load_2d_bf16(vlds, vbase + (size_t)kt * HDIM, HDIM, 32, HDIM, 32, HDIM);
            __builtin_amdgcn_s_wait_tensorcnt(0);
        }
        __syncthreads();  // tiles visible to all waves

        // ---- scores: 16 q-rows x 32 keys (K/V fragments from LDS) ------
        v8f s0 = {}, s1 = {};
        {
            v16bf bk0 = load_frag_rowmajor_bf16(&ktile[0], HDIM);            // keys 0-15, d 0-31
            v16bf bk1 = load_frag_rowmajor_bf16(&ktile[32], HDIM);           // keys 0-15, d 32-63
            s0 = wmma_bf16(aq0, bk0, s0);
            s0 = wmma_bf16(aq1, bk1, s0);
            v16bf bk2 = load_frag_rowmajor_bf16(&ktile[16 * HDIM], HDIM);    // keys 16-31
            v16bf bk3 = load_frag_rowmajor_bf16(&ktile[16 * HDIM + 32], HDIM);
            s1 = wmma_bf16(aq0, bk2, s1);
            s1 = wmma_bf16(aq1, bk3, s1);
        }

        // ---- mask + scale ----------------------------------------------
        float e0[8], e1[8], mx[8];
#pragma unroll
        for (int i = 0; i < 8; ++i) {
            const int qpos = q0 + grp * 8 + i;
            const int m0 = mask[(size_t)qpos * SEQ + kt + r];
            const int m1 = mask[(size_t)qpos * SEQ + kt + 16 + r];
            e0[i] = m0 ? s0[i] * scale : -1.0e30f;
            e1[i] = m1 ? s1[i] * scale : -1.0e30f;
            mx[i] = fmaxf(e0[i], e1[i]);
        }
        // row max over the 16 lanes that hold each row
#pragma unroll
        for (int off = 1; off < 16; off <<= 1)
#pragma unroll
            for (int i = 0; i < 8; ++i) mx[i] = fmaxf(mx[i], __shfl_xor(mx[i], off, 32));

        // ---- online softmax update -------------------------------------
        float psum[8], osc[8];
#pragma unroll
        for (int i = 0; i < 8; ++i) {
            const float mnew = fmaxf(mrow[i], mx[i]);
            osc[i] = __expf(mrow[i] - mnew);
            mrow[i] = mnew;
            const float p0 = __expf(e0[i] - mnew);
            const float p1 = __expf(e1[i] - mnew);
            p_lds[wave][(grp * 8 + i) * 32 + r] = (__bf16)p0;
            p_lds[wave][(grp * 8 + i) * 32 + 16 + r] = (__bf16)p1;
            psum[i] = p0 + p1;
        }
#pragma unroll
        for (int off = 1; off < 16; off <<= 1)
#pragma unroll
            for (int i = 0; i < 8; ++i) psum[i] += __shfl_xor(psum[i], off, 32);
#pragma unroll
        for (int i = 0; i < 8; ++i) lrow[i] = lrow[i] * osc[i] + psum[i];
#pragma unroll
        for (int nc = 0; nc < 4; ++nc)
#pragma unroll
            for (int i = 0; i < 8; ++i) oacc[nc][i] *= osc[i];

        // ---- P @ V: reload P from LDS in A-fragment layout -------------
        v16bf pa;
        {
            const __bf16* pl = &p_lds[wave][0];
#pragma unroll
            for (int j = 0; j < 8; ++j) {
                const int kb = frag_kbase(grp, j);
                pa[2 * j]     = pl[r * 32 + kb];
                pa[2 * j + 1] = pl[r * 32 + kb + 1];
            }
        }
#pragma unroll
        for (int nc = 0; nc < 4; ++nc) {
            v16bf bv = load_frag_kmajor_bf16(&vtile[nc * 16], HDIM);
            oacc[nc] = wmma_bf16(pa, bv, oacc[nc]);
        }
    }

    // epilogue: O / l  -> attn_out[n, q, h*64 + d]  (bf16, [B,S,E])
    __bf16* o = attn_out + ((size_t)n * SEQ + q0) * EMBED + h * HDIM;
#pragma unroll
    for (int nc = 0; nc < 4; ++nc)
#pragma unroll
        for (int i = 0; i < 8; ++i) {
            const int M = grp * 8 + i;
            o[(size_t)M * EMBED + nc * 16 + r] = (__bf16)(oacc[nc][i] / lrow[i]);
        }
}

// ---------------------------------------------------------------------------
// Kernel 3a: Wo fp32 -> bf16 (same [E][E] layout)
// ---------------------------------------------------------------------------
__global__ void cvt_bf16_kernel(const float* __restrict__ src, __bf16* __restrict__ dst,
                                int nelem) {
    const int idx = blockIdx.x * blockDim.x + threadIdx.x;
    if (idx < nelem) dst[idx] = (__bf16)src[idx];
}

// ---------------------------------------------------------------------------
// Kernel 3b: out = Attn @ Wo^T + bo   (M = B*S, N = K = 1024)
// All 8 waves of a block share one 64-col Wo tile (WGP$ reuse), 8 row tiles.
// ---------------------------------------------------------------------------
__global__ void outproj_kernel(const __bf16* __restrict__ X, const __bf16* __restrict__ Wb,
                               const float* __restrict__ bias, float* __restrict__ out) {
    const int wave = threadIdx.x >> 5;
    const int ct = blockIdx.x & 15;                    // 16 col tiles of 64
    const int rt = (blockIdx.x >> 4) * 8 + wave;       // 512 row tiles of 16
    const int row0 = rt * 16, col0 = ct * 64;
    const int lane = threadIdx.x & 31;
    const int r = lane & 15, grp = lane >> 4;

    v8f acc[4] = {};
    for (int kc = 0; kc < EMBED; kc += 32) {
        v16bf a = load_frag_rowmajor_bf16(X + (size_t)row0 * EMBED + kc, EMBED);
#pragma unroll
        for (int nc = 0; nc < 4; ++nc) {
            // B(k=c, n=e') = Wo[e', c]: N-major with ld = EMBED
            v16bf b = load_frag_rowmajor_bf16(Wb + (size_t)(col0 + nc * 16) * EMBED + kc,
                                              EMBED);
            acc[nc] = wmma_bf16(a, b, acc[nc]);
        }
    }
#pragma unroll
    for (int nc = 0; nc < 4; ++nc)
#pragma unroll
        for (int i = 0; i < 8; ++i) {
            const int M = row0 + grp * 8 + i;
            const int N = col0 + nc * 16 + r;
            out[(size_t)M * EMBED + N] = acc[nc][i] + bias[N];
        }
}

// ---------------------------------------------------------------------------
extern "C" void kernel_launch(void* const* d_in, const int* in_sizes, int n_in,
                              void* d_out, int out_size, void* d_ws, size_t ws_size,
                              hipStream_t stream) {
    const float* q    = (const float*)d_in[0];
    const float* k    = (const float*)d_in[1];
    const float* v    = (const float*)d_in[2];
    const int*   mask = (const int*)d_in[3];
    const float* Wq   = (const float*)d_in[4];
    const float* Wk   = (const float*)d_in[5];
    const float* Wv   = (const float*)d_in[6];
    const float* Wo   = (const float*)d_in[7];
    const float* bo   = (const float*)d_in[8];
    float* out = (float*)d_out;

    const size_t nproj = (size_t)BATCH * HEADS * SEQ * HDIM;  // 8M elements
    __bf16* Qp   = (__bf16*)d_ws;
    __bf16* Kp   = Qp + nproj;
    __bf16* Vp   = Kp + nproj;
    __bf16* Attn = Vp + nproj;                                 // B*S*E = 8M
    __bf16* Wob  = Attn + (size_t)BATCH * SEQ * EMBED;         // 1M

    const dim3 blk(256);
    // B*H*S/16 = 8192 wave-tiles / 8 waves per block = 1024 blocks
    proj_kernel<<<1024, blk, 0, stream>>>(q, Wq, Qp);
    proj_kernel<<<1024, blk, 0, stream>>>(k, Wk, Kp);
    proj_kernel<<<1024, blk, 0, stream>>>(v, Wv, Vp);
    cvt_bf16_kernel<<<(EMBED * EMBED + 255) / 256, blk, 0, stream>>>(Wo, Wob,
                                                                    EMBED * EMBED);
    // B*H*(S/128) = 1024 blocks
    attn_kernel<<<1024, blk, 0, stream>>>(Qp, Kp, Vp, mask, Attn);
    // 16 col tiles x 64 row-tile groups = 1024 blocks
    outproj_kernel<<<1024, blk, 0, stream>>>(Attn, Wob, bo, out);
}
